// MCIB_24215025614946
// MI455X (gfx1250) — compile-verified, matrix operand order checked by reading, not run on hardware
//
#include <hip/hip_runtime.h>
#include <hip/hip_bf16.h>
#include <stdint.h>

// ---------------------------------------------------------------------------
// Types for CDNA5 WMMA (wave32): A/B = 16 bf16 per lane, C/D = 8 f32 per lane
// ---------------------------------------------------------------------------
typedef __bf16 bf16_t;
typedef __bf16 v16bf __attribute__((ext_vector_type(16)));
typedef float  v8f   __attribute__((ext_vector_type(8)));

// Tensor Data Mover availability (device pass only; host pass falls back)
#if defined(__has_builtin)
#  if __has_builtin(__builtin_amdgcn_tensor_load_to_lds) && \
      __has_builtin(__builtin_amdgcn_s_wait_tensorcnt)
#    define USE_TDM 1
#  endif
#endif

__device__ __forceinline__ v8f wmma_bf16(v16bf a, v16bf b, v8f c) {
  // D = A(16x32 bf16) x B(32x16 bf16) + C(16x16 f32)
  return __builtin_amdgcn_wmma_f32_16x16x32_bf16(
      /*neg_a=*/false, a, /*neg_b=*/false, b,
      /*c_mod=*/(short)0, c, /*reuse_a=*/false, /*reuse_b=*/false);
}

// Load one 16x32 bf16 fragment (A-layout; also used for B where "row" = N col
// and per-lane elements run along K). Lanes 0-15 -> K {0-7,16-23},
// lanes 16-31 -> K {8-15,24-31}; two ds_load_b128 per lane.
__device__ __forceinline__ v16bf load_frag_lds(const bf16_t* base, int row0,
                                               int stride, int k0, int lane) {
  int r  = lane & 15;
  int hl = lane >> 4;
  const bf16_t* p = base + (size_t)(row0 + r) * stride + k0 + hl * 8;
  union { v16bf v; uint4 q[2]; } u;
  u.q[0] = *(const uint4*)(p);
  u.q[1] = *(const uint4*)(p + 16);
  return u.v;
}

// ---------------------------------------------------------------------------
// Elementwise f32 -> bf16 conversion, 8 elements per thread (b128 traffic)
// ---------------------------------------------------------------------------
__global__ void cvt_bf16_kernel(const float* __restrict__ in,
                                bf16_t* __restrict__ out, int n8) {
  int i = blockIdx.x * blockDim.x + threadIdx.x;
  if (i >= n8) return;
  const float4* src = (const float4*)in + i * 2;
  float4 a = src[0], b = src[1];
  union { uint4 q; bf16_t h[8]; } u;
  u.h[0] = (bf16_t)a.x; u.h[1] = (bf16_t)a.y;
  u.h[2] = (bf16_t)a.z; u.h[3] = (bf16_t)a.w;
  u.h[4] = (bf16_t)b.x; u.h[5] = (bf16_t)b.y;
  u.h[6] = (bf16_t)b.z; u.h[7] = (bf16_t)b.w;
  ((uint4*)out)[i] = u.q;
}

// ---------------------------------------------------------------------------
// Tiled WMMA GEMM:  C[M,N] = A[M,K](bf16) * W[N,K]^T(bf16) + bias[N]
// 256 threads = 8 waves (4 along M x 2 along N); block tile 128x128,
// wave tile 32x64 (2x4 WMMA), k-step 32, double-buffered LDS with
// register-staged global loads (one barrier per k-step).
// OMODE bit0: write f32 output, bit1: write bf16 output (compile-time).
// ---------------------------------------------------------------------------
template <int OMODE>
__global__ __launch_bounds__(256, 1) void gemm_bias_kernel(
    const bf16_t* __restrict__ A, const bf16_t* __restrict__ W,
    const float* __restrict__ bias,
    float* __restrict__ outF, bf16_t* __restrict__ outB,
    int M, int N, int K, float qscale, int qcols) {
  __shared__ __align__(16) bf16_t As[2][128 * 32];
  __shared__ __align__(16) bf16_t Ws[2][128 * 32];

  int tid  = threadIdx.x;
  int lane = tid & 31;
  int wave = tid >> 5;
  int wm = wave >> 1;      // 0..3 (M)
  int wn = wave & 1;       // 0..1 (N)
  int m0 = blockIdx.x * 128;
  int n0 = blockIdx.y * 128;

  // Staging role: thread -> row r (0..127), 16-elem column chunk
  int r   = tid >> 1;
  int cst = (tid & 1) * 16;
  const bf16_t* Arow = A + (size_t)(m0 + r) * K + cst;
  const bf16_t* Wrow = W + (size_t)(n0 + r) * K + cst;

  v8f acc[2][4] = {};
  const int NIT = K / 32;

  uint4 ra0, ra1, rw0, rw1;
  // Prologue: stage k-chunk 0 into buffer 0
  ra0 = *(const uint4*)(Arow);      ra1 = *(const uint4*)(Arow + 8);
  rw0 = *(const uint4*)(Wrow);      rw1 = *(const uint4*)(Wrow + 8);
  *(uint4*)(As[0] + r * 32 + cst)     = ra0;
  *(uint4*)(As[0] + r * 32 + cst + 8) = ra1;
  *(uint4*)(Ws[0] + r * 32 + cst)     = rw0;
  *(uint4*)(Ws[0] + r * 32 + cst + 8) = rw1;
  __syncthreads();

  for (int it = 0; it < NIT; ++it) {
    int buf = it & 1;
    bool more = (it + 1) < NIT;
    if (more) {
      int kk = (it + 1) * 32;
      ra0 = *(const uint4*)(Arow + kk);     ra1 = *(const uint4*)(Arow + kk + 8);
      rw0 = *(const uint4*)(Wrow + kk);     rw1 = *(const uint4*)(Wrow + kk + 8);
      if (kk + 64 < K) {   // software prefetch two tiles ahead
        __builtin_prefetch(Arow + kk + 64, 0, 3);
        __builtin_prefetch(Wrow + kk + 64, 0, 3);
      }
    }

    v16bf af[2], wf[4];
    for (int i = 0; i < 2; i++)
      af[i] = load_frag_lds(As[buf], wm * 32 + i * 16, 32, 0, lane);
    for (int j = 0; j < 4; j++)
      wf[j] = load_frag_lds(Ws[buf], wn * 64 + j * 16, 32, 0, lane);
    for (int i = 0; i < 2; i++)
      for (int j = 0; j < 4; j++)
        acc[i][j] = wmma_bf16(af[i], wf[j], acc[i][j]);

    if (more) {   // publish next buffer after compute issued
      *(uint4*)(As[buf ^ 1] + r * 32 + cst)     = ra0;
      *(uint4*)(As[buf ^ 1] + r * 32 + cst + 8) = ra1;
      *(uint4*)(Ws[buf ^ 1] + r * 32 + cst)     = rw0;
      *(uint4*)(Ws[buf ^ 1] + r * 32 + cst + 8) = rw1;
    }
    __syncthreads();
  }

  // Epilogue: C layout -> lane holds col n = lane%16, rows 8*(lane/16)+e
  int cr = lane & 15;
  int hl = lane >> 4;
  for (int i = 0; i < 2; i++) {
    int row0 = m0 + wm * 32 + i * 16 + hl * 8;
    for (int j = 0; j < 4; j++) {
      int col = n0 + wn * 64 + j * 16 + cr;
      float bv  = bias[col];
      float scl = (col < qcols) ? qscale : 1.0f;
      for (int e = 0; e < 8; e++) {
        float v = (acc[i][j][e] + bv) * scl;
        size_t idx = (size_t)(row0 + e) * N + col;
        if (OMODE & 1) outF[idx] = v;
        if (OMODE & 2) outB[idx] = (bf16_t)v;
      }
    }
  }
}

// ---------------------------------------------------------------------------
// TDM: async-load one 32(keys) x 256(d) bf16 tile (row stride 1536 elems)
// from global into LDS.  D# layout per CDNA5 ISA ch.8 (2D tensor, type=2).
// ---------------------------------------------------------------------------
__device__ __forceinline__ void tdm_load_k_tile(bf16_t* ldsDst,
                                                const bf16_t* gsrc) {
#if defined(USE_TDM)
  typedef unsigned int u32x4 __attribute__((ext_vector_type(4)));
  typedef int i32x4 __attribute__((ext_vector_type(4)));
  typedef int i32x8 __attribute__((ext_vector_type(8)));
  uint64_t ga = (uint64_t)(uintptr_t)gsrc;
  u32x4 g0;
  g0[0] = 1u;                                   // count=1 (valid descriptor)
  g0[1] = (uint32_t)(uintptr_t)ldsDst;          // lds_addr
  g0[2] = (uint32_t)ga;                         // global_addr[31:0]
  g0[3] = (uint32_t)((ga >> 32) & 0x01FFFFFFu)  // global_addr[56:32]
          | (2u << 30);                         // type = 2 ("image")
  i32x8 g1;
  g1[0] = 0x00010000;            // workgroup_mask=0, data_size=1 (2 bytes)
  g1[1] = (int)(256u << 16);     // tensor_dim0 = 256 (d)
  g1[2] = (int)(2048u << 16);    // tensor_dim1 = 2048 (keys)
  g1[3] = (int)(256u << 16);     // tile_dim0 = 256
  g1[4] = 32;                    // tile_dim1 = 32, tile_dim2 = 0
  g1[5] = 1536;                  // tensor_dim0_stride = row pitch (elems)
  g1[6] = 0;
  g1[7] = 0;
  i32x4 z4 = {0, 0, 0, 0};
#if __clang_major__ >= 23
  i32x8 z8 = {0, 0, 0, 0, 0, 0, 0, 0};
  __builtin_amdgcn_tensor_load_to_lds(g0, g1, z4, z4, z8, 0);
#else
  __builtin_amdgcn_tensor_load_to_lds(g0, g1, z4, z4, 0);
#endif
#else
  (void)ldsDst; (void)gsrc;
#endif
}

// ---------------------------------------------------------------------------
// Flash attention (H=2, hd=256, S=2048). qkv buffers: [B*S][1536] bf16 with
// q|k|v packed (q pre-scaled by 1/16). Output [B*S][512] bf16.
// 128 threads = 4 waves x 16 q-rows; 32-key K/V tiles double-buffered in LDS.
// K staged via TDM (fallback: ds stores); V staged transposed for PV B-frags.
// ---------------------------------------------------------------------------
__global__ __launch_bounds__(128, 1) void flash_attn_kernel(
    const bf16_t* __restrict__ Qbuf, const bf16_t* __restrict__ KVbuf,
    bf16_t* __restrict__ Obuf) {
  const int S = 2048, HD = 256, LDQKV = 1536, LDO = 512;

  __shared__ __align__(16) bf16_t Ks[2][32 * 256];    // [key][d]
  __shared__ __align__(16) bf16_t VTs[2][256 * 32];   // [d][key]
  __shared__ __align__(16) bf16_t Qs[4][16 * 256];    // per-wave Q tile
  __shared__ __align__(16) bf16_t Ps[4][16 * 32];     // per-wave P restage

  int tid  = threadIdx.x;
  int lane = tid & 31;
  int wave = tid >> 5;
  int bh = blockIdx.y;          // 0..15
  int b  = bh >> 1;
  int h  = bh & 1;
  size_t rowbase = (size_t)b * S;
  int q0 = blockIdx.x * 64 + wave * 16;

  // Stage this wave's Q tile (16x256) into LDS
  {
    const bf16_t* qsrc = Qbuf + (rowbase + q0) * LDQKV + h * HD;
    int qr = lane >> 1;
    int c0 = (lane & 1) * 128;
    for (int c = 0; c < 128; c += 8)
      *(uint4*)(&Qs[wave][qr * 256 + c0 + c]) =
          *(const uint4*)(qsrc + (size_t)qr * LDQKV + c0 + c);
  }

  // Staging roles for K/V tiles
  int sr = tid >> 2;            // key 0..31
  int sc = (tid & 3) * 64;      // 64 d-elements per thread
  const bf16_t* kbase = KVbuf + rowbase * LDQKV + 512 + h * HD;
  const bf16_t* vbase = KVbuf + rowbase * LDQKV + 1024 + h * HD;

  auto stage_kv = [&](int buf, int kt) {
#ifndef USE_TDM
    const bf16_t* ksrc = kbase + (size_t)(kt * 32 + sr) * LDQKV + sc;
    for (int c = 0; c < 64; c += 8)
      *(uint4*)(&Ks[buf][sr * 256 + sc + c]) = *(const uint4*)(ksrc + c);
#endif
    // V: vectorized global reads (b128), transpose via b16 LDS scatter
    const bf16_t* vsrc = vbase + (size_t)(kt * 32 + sr) * LDQKV + sc;
    for (int c = 0; c < 64; c += 8) {
      union { uint4 q; bf16_t h[8]; } vv;
      vv.q = *(const uint4*)(vsrc + c);
      for (int e = 0; e < 8; e++)
        VTs[buf][(sc + c + e) * 32 + sr] = vv.h[e];
    }
  };

  float mrow[8], lrow[8];
  for (int e = 0; e < 8; e++) { mrow[e] = -1e30f; lrow[e] = 0.0f; }
  v8f Oacc[16] = {};   // 16 q-rows x 256 d in f32 C layout

  // Prologue: stage key tile 0 into buffer 0
#ifdef USE_TDM
  if (wave == 0) tdm_load_k_tile(&Ks[0][0], kbase);
#endif
  stage_kv(0, 0);
#ifdef USE_TDM
  if (wave == 0) __builtin_amdgcn_s_wait_tensorcnt(0);
#endif
  __syncthreads();

  const int NT = S / 32;
  for (int kt = 0; kt < NT; kt++) {
    int cur = kt & 1;
    bool more = (kt + 1) < NT;
    if (more) {
#ifdef USE_TDM
      if (wave == 0)
        tdm_load_k_tile(&Ks[cur ^ 1][0],
                        kbase + (size_t)(kt + 1) * 32 * LDQKV);
#endif
      stage_kv(cur ^ 1, kt + 1);
    }

    // Scores: two 16-key subtiles, accumulate over hd=256 (8 k-chunks)
    v8f s0 = {}, s1 = {};
    for (int dc = 0; dc < 8; dc++) {
      v16bf qf  = load_frag_lds(Qs[wave], 0, 256, dc * 32, lane);
      v16bf kf0 = load_frag_lds(Ks[cur], 0,  256, dc * 32, lane);
      v16bf kf1 = load_frag_lds(Ks[cur], 16, 256, dc * 32, lane);
      s0 = wmma_bf16(qf, kf0, s0);
      s1 = wmma_bf16(qf, kf1, s1);
    }

    // Online softmax in C layout (row r spans 16 lanes of one half)
    float alpha[8];
    v8f p0, p1;
    for (int e = 0; e < 8; e++) {
      float v = fmaxf(s0[e], s1[e]);
      for (int m = 1; m < 16; m <<= 1) v = fmaxf(v, __shfl_xor(v, m, 32));
      float nm = fmaxf(mrow[e], v);
      alpha[e] = __expf(mrow[e] - nm);
      p0[e] = __expf(s0[e] - nm);
      p1[e] = __expf(s1[e] - nm);
      float rs = p0[e] + p1[e];
      for (int m = 1; m < 16; m <<= 1) rs += __shfl_xor(rs, m, 32);
      lrow[e] = lrow[e] * alpha[e] + rs;
      mrow[e] = nm;
    }
    for (int c = 0; c < 16; c++)
      for (int e = 0; e < 8; e++) Oacc[c][e] *= alpha[e];

    // Restage P (16x32) row-major so it reloads as an A-fragment
    {
      int pr = lane & 15;
      int hl = lane >> 4;
      for (int e = 0; e < 8; e++) {
        int row = hl * 8 + e;
        Ps[wave][row * 32 + pr]      = (bf16_t)p0[e];
        Ps[wave][row * 32 + 16 + pr] = (bf16_t)p1[e];
      }
    }

    // PV: A = P (16x32 keys), B = V^T chunks (lane = d col, elems over keys)
    v16bf pf = load_frag_lds(Ps[wave], 0, 32, 0, lane);
    for (int dc = 0; dc < 16; dc++) {
      v16bf vf = load_frag_lds(VTs[cur], dc * 16, 32, 0, lane);
      Oacc[dc] = wmma_bf16(pf, vf, Oacc[dc]);
    }

#ifdef USE_TDM
    if (more && wave == 0) __builtin_amdgcn_s_wait_tensorcnt(0);
#endif
    __syncthreads();
  }

  // Normalize and store
  {
    int cr = lane & 15;
    int hl = lane >> 4;
    for (int dc = 0; dc < 16; dc++)
      for (int e = 0; e < 8; e++) {
        int row = q0 + hl * 8 + e;
        int col = h * HD + dc * 16 + cr;
        Obuf[(rowbase + row) * LDO + col] = (bf16_t)(Oacc[dc][e] / lrow[e]);
      }
  }
}

// ---------------------------------------------------------------------------
// gate = sigmoid(<rgb_p, event_p>); out = gate*eattn + (1-gate)*rattn
//        + 0.5*(rgb_p + event_p).  One wave per row, float4 vectorized.
// ---------------------------------------------------------------------------
__global__ __launch_bounds__(256) void fuse_kernel(
    const float* __restrict__ rgb_p, const float* __restrict__ event_p,
    const float* __restrict__ eattn, const float* __restrict__ rattn,
    float* __restrict__ out, int D) {
  int lane = threadIdx.x & 31;
  int wave = threadIdx.x >> 5;
  size_t row = (size_t)blockIdx.x * 8 + wave;
  int D4 = D >> 2;
  const float4* rp = (const float4*)(rgb_p + row * D);
  const float4* ep = (const float4*)(event_p + row * D);
  float dot = 0.0f;
  for (int c = lane; c < D4; c += 32) {
    float4 a = rp[c], b = ep[c];
    dot += a.x * b.x + a.y * b.y + a.z * b.z + a.w * b.w;
  }
  for (int m = 1; m < 32; m <<= 1) dot += __shfl_xor(dot, m, 32);
  float gate = 1.0f / (1.0f + __expf(-dot));
  const float4* ea = (const float4*)(eattn + row * D);
  const float4* ra = (const float4*)(rattn + row * D);
  float4* o = (float4*)(out + row * D);
  for (int c = lane; c < D4; c += 32) {
    float4 a = rp[c], b = ep[c], x = ea[c], y = ra[c], z;
    z.x = gate * x.x + (1.0f - gate) * y.x + 0.5f * (a.x + b.x);
    z.y = gate * x.y + (1.0f - gate) * y.y + 0.5f * (a.y + b.y);
    z.z = gate * x.z + (1.0f - gate) * y.z + 0.5f * (a.z + b.z);
    z.w = gate * x.w + (1.0f - gate) * y.w + 0.5f * (a.w + b.w);
    o[c] = z;
  }
}

// ---------------------------------------------------------------------------
// Host-side orchestration
// ---------------------------------------------------------------------------
extern "C" void kernel_launch(void* const* d_in, const int* in_sizes, int n_in,
                              void* d_out, int out_size, void* d_ws,
                              size_t ws_size, hipStream_t stream) {
  (void)in_sizes; (void)n_in; (void)out_size; (void)ws_size;
  const int B = 8, S = 2048, D = 512, M = B * S;   // per reference

  const float* rgb     = (const float*)d_in[0];
  const float* event_  = (const float*)d_in[1];
  const float* w_rgb   = (const float*)d_in[2];
  const float* b_rgb   = (const float*)d_in[3];
  const float* w_event = (const float*)d_in[4];
  const float* b_event = (const float*)d_in[5];
  const float* w_in    = (const float*)d_in[6];
  const float* b_in    = (const float*)d_in[7];
  const float* w_out   = (const float*)d_in[8];
  const float* b_out   = (const float*)d_in[9];
  float* out = (float*)d_out;

  // Deterministic workspace carve-up (~335 MB total)
  char* ws = (char*)d_ws;
  size_t off = 0;
  auto alloc = [&](size_t bytes) {
    void* p = ws + off;
    off = (off + bytes + 255) & ~(size_t)255;
    return p;
  };
  bf16_t* rgb_bf    = (bf16_t*)alloc((size_t)M * D * 2);
  bf16_t* event_bf  = (bf16_t*)alloc((size_t)M * D * 2);
  bf16_t* w_rgb_bf  = (bf16_t*)alloc((size_t)D * D * 2);
  bf16_t* w_event_bf= (bf16_t*)alloc((size_t)D * D * 2);
  bf16_t* w_in_bf   = (bf16_t*)alloc((size_t)3 * D * D * 2);
  bf16_t* w_out_bf  = (bf16_t*)alloc((size_t)D * D * 2);
  float*  rgb_p_f   = (float*) alloc((size_t)M * D * 4);
  float*  event_p_f = (float*) alloc((size_t)M * D * 4);
  bf16_t* rgb_p_bf  = (bf16_t*)alloc((size_t)M * D * 2);
  bf16_t* event_p_bf= (bf16_t*)alloc((size_t)M * D * 2);
  bf16_t* qkv_rgb   = (bf16_t*)alloc((size_t)M * 3 * D * 2);
  bf16_t* qkv_event = (bf16_t*)alloc((size_t)M * 3 * D * 2);
  bf16_t* attnE_bf  = (bf16_t*)alloc((size_t)M * D * 2);
  bf16_t* attnR_bf  = (bf16_t*)alloc((size_t)M * D * 2);
  float*  eattn_f   = (float*) alloc((size_t)M * D * 4);
  float*  rattn_f   = (float*) alloc((size_t)M * D * 4);

  auto cvt = [&](const float* src, bf16_t* dst, int n) {
    int n8 = n / 8;
    cvt_bf16_kernel<<<(n8 + 255) / 256, 256, 0, stream>>>(src, dst, n8);
  };
  cvt(rgb, rgb_bf, M * D);
  cvt(event_, event_bf, M * D);
  cvt(w_rgb, w_rgb_bf, D * D);
  cvt(w_event, w_event_bf, D * D);
  cvt(w_in, w_in_bf, 3 * D * D);
  cvt(w_out, w_out_bf, D * D);

  const float qscale = 0.0625f;  // 1/sqrt(256)
  dim3 blk(256);

  // Input projections: f32 (gate/residual) + bf16 (next GEMMs)   OMODE=3
  gemm_bias_kernel<3><<<dim3(M / 128, D / 128), blk, 0, stream>>>(
      rgb_bf, w_rgb_bf, b_rgb, rgb_p_f, rgb_p_bf, M, D, D, 1.0f, 0);
  gemm_bias_kernel<3><<<dim3(M / 128, D / 128), blk, 0, stream>>>(
      event_bf, w_event_bf, b_event, event_p_f, event_p_bf, M, D, D, 1.0f, 0);

  // Packed QKV projections (q columns scaled by 1/sqrt(hd))      OMODE=2
  gemm_bias_kernel<2><<<dim3(M / 128, (3 * D) / 128), blk, 0, stream>>>(
      rgb_p_bf, w_in_bf, b_in, nullptr, qkv_rgb, M, 3 * D, D, qscale, D);
  gemm_bias_kernel<2><<<dim3(M / 128, (3 * D) / 128), blk, 0, stream>>>(
      event_p_bf, w_in_bf, b_in, nullptr, qkv_event, M, 3 * D, D, qscale, D);

  // Cross attention, both directions (grid: 32 q-blocks x B*H)
  flash_attn_kernel<<<dim3(S / 64, B * 2), dim3(128), 0, stream>>>(
      qkv_rgb, qkv_event, attnE_bf);   // q=rgb_p, kv=event_p -> event_attn
  flash_attn_kernel<<<dim3(S / 64, B * 2), dim3(128), 0, stream>>>(
      qkv_event, qkv_rgb, attnR_bf);   // q=event_p, kv=rgb_p -> rgb_attn

  // Output projections                                           OMODE=1
  gemm_bias_kernel<1><<<dim3(M / 128, D / 128), blk, 0, stream>>>(
      attnE_bf, w_out_bf, b_out, eattn_f, nullptr, M, D, D, 1.0f, 0);
  gemm_bias_kernel<1><<<dim3(M / 128, D / 128), blk, 0, stream>>>(
      attnR_bf, w_out_bf, b_out, rattn_f, nullptr, M, D, D, 1.0f, 0);

  // Gate + fuse + residual
  fuse_kernel<<<dim3(M / 8), blk, 0, stream>>>(rgb_p_f, event_p_f, eattn_f,
                                               rattn_f, out, D);
}
